// BitNetV3Attention_10419590660565
// MI455X (gfx1250) — compile-verified, matrix operand order checked by reading, not run on hardware
//
#include <hip/hip_runtime.h>
#include <hip/hip_bf16.h>

typedef __attribute__((ext_vector_type(16))) _Float16 v16h;
typedef __attribute__((ext_vector_type(8)))  _Float16 v8h;
typedef __attribute__((ext_vector_type(2)))  _Float16 v2h;
typedef __attribute__((ext_vector_type(8)))  float    v8f;

// Problem constants (match reference harness)
#define Bc  2
#define Sc  2048
#define Hc  16
#define DHc 128
#define Dc  2048          // Hc*DHc
#define Mc  (Bc*Sc)       // 4096 tokens

#define WMMA_F16(a, b, c) \
    __builtin_amdgcn_wmma_f32_16x16x32_f16(false, (a), false, (b), (short)0, (c), false, false)

// Load a 16x32 f16 fragment (A or B operand, symmetric per-lane layout):
//   lane L: row = base_row + (L&15), K-halves at col + (L>>4)*8 and col + 16 + (L>>4)*8
__device__ __forceinline__ v16h load_frag16(const _Float16* __restrict__ p, size_t ld,
                                            int row, int col, int lane) {
    const int r = lane & 15;
    const int hs = lane >> 4;
    const _Float16* a = p + (size_t)(row + r) * ld + (size_t)(col + hs * 8);
    union { v16h v; v8h h[2]; } u;
    u.h[0] = *(const v8h*)(a);
    u.h[1] = *(const v8h*)(a + 16);
    return u.v;
}

// -------------------- fp32 -> fp16 convert --------------------
__global__ void cvt_f32_f16(const float* __restrict__ in, _Float16* __restrict__ out, int n) {
    for (int i = blockIdx.x * blockDim.x + threadIdx.x; i < n; i += gridDim.x * blockDim.x)
        out[i] = (_Float16)in[i];
}

// -------------------- Y = X @ W^T  (X:[M,K] f16, W:[N,K] f16) --------------------
// Each wave computes a 32x64 output tile (2x4 accumulators). K-loop in steps of 32
// with ping-pong double-buffered fragments (unroll-2 steady state): no register
// rotation copies, loads for step s+1 overlap the WMMAs of step s.
// mode 0: write f16 as Q/K layout  [B,H,S,DH]
// mode 2: write f16 as V^T layout  [B,H,DH,S]
// mode 3: write f32 row-major      [M,N]   (final projection -> d_out)
__global__ void __launch_bounds__(128)
gemm_xwt(const _Float16* __restrict__ X, const _Float16* __restrict__ W,
         void* __restrict__ Yout, int mode) {
    const int Mi = Mc, Ni = Dc, Ki = Dc;
    const int tid  = threadIdx.x;
    const int lane = tid & 31;
    const int wid  = blockIdx.x * (blockDim.x >> 5) + (tid >> 5);
    const int tilesM = Mi / 32;                 // 128
    const int tm = wid % tilesM;
    const int tn = wid / tilesM;                // 0..31
    const int row0 = tm * 32;
    const int col0 = tn * 64;

    v8f acc[2][4] = {};
    v16h aA[2], bA[4];      // buffer set A
    v16h aB[2], bB[4];      // buffer set B

    auto LOAD = [&](v16h* aa, v16h* bb, int k) {
        aa[0] = load_frag16(X, (size_t)Ki, row0,      k, lane);
        aa[1] = load_frag16(X, (size_t)Ki, row0 + 16, k, lane);
        bb[0] = load_frag16(W, (size_t)Ki, col0,      k, lane);
        bb[1] = load_frag16(W, (size_t)Ki, col0 + 16, k, lane);
        bb[2] = load_frag16(W, (size_t)Ki, col0 + 32, k, lane);
        bb[3] = load_frag16(W, (size_t)Ki, col0 + 48, k, lane);
    };
    auto COMP = [&](const v16h* aa, const v16h* bb) {
#pragma unroll
        for (int i = 0; i < 2; ++i)
#pragma unroll
            for (int j = 0; j < 4; ++j)
                acc[i][j] = WMMA_F16(aa[i], bb[j], acc[i][j]);
    };

    const int nsteps = Ki / 32;         // 64 (even)
    LOAD(aA, bA, 0);
#pragma unroll 1
    for (int s = 1; s <= nsteps - 3; s += 2) {
        LOAD(aB, bB, s * 32);           // prefetch odd step
        COMP(aA, bA);                   // compute even step
        LOAD(aA, bA, (s + 1) * 32);     // prefetch next even step
        COMP(aB, bB);                   // compute odd step
    }
    LOAD(aB, bB, (nsteps - 1) * 32);
    COMP(aA, bA);
    COMP(aB, bB);

    const int r15 = lane & 15;
    const int hs  = lane >> 4;
#pragma unroll
    for (int i = 0; i < 2; ++i) {
#pragma unroll
        for (int j = 0; j < 4; ++j) {
#pragma unroll
            for (int r = 0; r < 8; ++r) {
                const int m = row0 + i * 16 + r + 8 * hs;   // token row
                const int n = col0 + j * 16 + r15;          // output feature
                const float val = acc[i][j][r];
                if (mode == 3) {
                    ((float*)Yout)[(size_t)m * Ni + n] = val;
                } else {
                    const int b = m >> 11, s = m & (Sc - 1);
                    const int h = n >> 7,  d = n & (DHc - 1);
                    if (mode == 2) {   // V^T: [B,H,DH,S]
                        ((_Float16*)Yout)[((((size_t)b * Hc + h) * DHc) + d) * Sc + s] = (_Float16)val;
                    } else {           // Q/K: [B,H,S,DH]
                        ((_Float16*)Yout)[((((size_t)b * Hc + h) * Sc) + s) * DHc + d] = (_Float16)val;
                    }
                }
            }
        }
    }
}

// -------------------- Flash attention (causal + padding mask) --------------------
// One wave per (b,h, 16-query tile). Scores computed transposed: S^T = K_tile @ Q^T,
// so queries live across lanes and keys across accumulator VGPRs; P^T packs directly
// into the B fragment of O^T += V^T @ P^T with zero cross-lane traffic.
// All K and V fragment loads for an iteration are issued up front so the V loads
// complete under the score WMMAs + softmax VALU work.
__global__ void __launch_bounds__(128)
flash_attn(const _Float16* __restrict__ Q, const _Float16* __restrict__ K,
           const _Float16* __restrict__ Vt, const int* __restrict__ amask,
           _Float16* __restrict__ O) {
    const int tid  = threadIdx.x;
    const int lane = tid & 31;
    const int wid  = blockIdx.x * (blockDim.x >> 5) + (tid >> 5);
    const int qt = wid % (Sc / 16);     // query tile
    const int bh = wid / (Sc / 16);     // batch*head
    const int b = bh >> 4, h = bh & 15;
    const int q0 = qt * 16;
    const int qlane = lane & 15;
    const int hs    = lane >> 4;
    const int qg = q0 + qlane;          // this lane's global query index

    const _Float16* Qb = Q  + (size_t)bh * Sc * DHc;
    const _Float16* Kb = K  + (size_t)bh * Sc * DHc;
    const _Float16* Vb = Vt + (size_t)bh * DHc * Sc;

    // Q as B-fragments (held in registers across the whole key loop)
    v16h qf[4];
#pragma unroll
    for (int c = 0; c < 4; ++c) qf[c] = load_frag16(Qb, DHc, q0, c * 32, lane);

    v8f o[8] = {};                       // O^T accumulators: 8 dh-tiles x (16x16)
    float mval = -1e30f, ssum = 0.f;
    const float sc = 0.08838834764831845f;   // 1/sqrt(128)

    const int niter = q0 / 32 + 1;       // causal: keys 0 .. q0+15
#pragma unroll 1
    for (int it = 0; it < niter; ++it) {
        const int kb = it * 32;

        // issue ALL loads for this iteration first: 8 K frags + 8 V frags
        v16h kf[8];
#pragma unroll
        for (int c = 0; c < 4; ++c) {
            kf[c]     = load_frag16(Kb, DHc, kb,      c * 32, lane);
            kf[4 + c] = load_frag16(Kb, DHc, kb + 16, c * 32, lane);
        }
        v16h vf[8];
#pragma unroll
        for (int t = 0; t < 8; ++t) vf[t] = load_frag16(Vb, (size_t)Sc, t * 16, kb, lane);

        // S^T tiles: c0 = keys kb..kb+15, c1 = keys kb+16..kb+31 (queries across lanes)
        v8f c0 = {}, c1 = {};
#pragma unroll
        for (int c = 0; c < 4; ++c) {
            c0 = WMMA_F16(kf[c],     qf[c], c0);
            c1 = WMMA_F16(kf[4 + c], qf[c], c1);
        }

        // scale + causal + padding mask (mask value for key kb+l held by lane l)
        const int mv = amask[b * Sc + kb + lane];
#pragma unroll
        for (int r = 0; r < 8; ++r) {
            const int key0 = kb + r + 8 * hs;
            const int key1 = key0 + 16;
            const int m0 = __shfl(mv, key0 - kb, 32);
            const int m1 = __shfl(mv, key1 - kb, 32);
            c0[r] = (key0 > qg || m0 == 0) ? -1e30f : c0[r] * sc;
            c1[r] = (key1 > qg || m1 == 0) ? -1e30f : c1[r] * sc;
        }

        // online softmax (per-lane stats; merge the two key-halves across lanes)
        float tm = -1e30f;
#pragma unroll
        for (int r = 0; r < 8; ++r) tm = fmaxf(tm, fmaxf(c0[r], c1[r]));
        tm = fmaxf(tm, __shfl_xor(tm, 16, 32));
        const float mnew = fmaxf(mval, tm);
        const float corr = __expf(mval - mnew);

        float lsum = 0.f;
        v16h pf;                          // P^T as B fragment: lane-local repack
#pragma unroll
        for (int r = 0; r < 8; ++r) {
            const float e0 = __expf(c0[r] - mnew);
            const float e1 = __expf(c1[r] - mnew);
            lsum += e0 + e1;
            pf[r]     = (_Float16)e0;
            pf[r + 8] = (_Float16)e1;
        }
        lsum += __shfl_xor(lsum, 16, 32);
        ssum = ssum * corr + lsum;
        mval = mnew;

#pragma unroll
        for (int t = 0; t < 8; ++t) {
#pragma unroll
            for (int i = 0; i < 8; ++i) o[t][i] *= corr;
        }
        // O^T += V^T @ P^T   (A = V^T dh-tile, loads already in flight/complete)
#pragma unroll
        for (int t = 0; t < 8; ++t) o[t] = WMMA_F16(vf[t], pf, o[t]);
    }

    // normalize and write O as [B*S, D] f16 (token-major for the final GEMM)
    const float rinv = 1.f / ssum;
    const size_t rowoff = ((size_t)(b * Sc + q0 + qlane)) * Dc + (size_t)h * DHc;
#pragma unroll
    for (int t = 0; t < 8; ++t) {
        const int dhb = t * 16 + 8 * hs;
#pragma unroll
        for (int r = 0; r < 8; r += 2) {
            v2h pv = { (_Float16)(o[t][r] * rinv), (_Float16)(o[t][r + 1] * rinv) };
            *(v2h*)(O + rowoff + dhb + r) = pv;
        }
    }
}

// -------------------- host-side launch --------------------
extern "C" void kernel_launch(void* const* d_in, const int* in_sizes, int n_in,
                              void* d_out, int out_size, void* d_ws, size_t ws_size,
                              hipStream_t stream) {
    const float* hidden = (const float*)d_in[0];
    const int*   amask  = (const int*)d_in[1];
    const float* Wq     = (const float*)d_in[2];
    const float* Wk     = (const float*)d_in[3];
    const float* Wv     = (const float*)d_in[4];
    const float* Wo     = (const float*)d_in[5];
    float* out = (float*)d_out;

    char* ws = (char*)d_ws;
    size_t off = 0;
    const size_t szX = (size_t)Mc * Dc * sizeof(_Float16);     // 16 MB
    const size_t szW = (size_t)Dc * Dc * sizeof(_Float16);     //  8 MB each
    _Float16* Xh   = (_Float16*)(ws + off); off += szX;
    _Float16* Wq16 = (_Float16*)(ws + off); off += szW;
    _Float16* Wk16 = (_Float16*)(ws + off); off += szW;
    _Float16* Wv16 = (_Float16*)(ws + off); off += szW;
    _Float16* Wo16 = (_Float16*)(ws + off); off += szW;
    _Float16* Qb   = (_Float16*)(ws + off); off += szX;        // [B,H,S,DH]
    _Float16* Kb   = (_Float16*)(ws + off); off += szX;        // [B,H,S,DH]
    _Float16* Vtb  = (_Float16*)(ws + off); off += szX;        // [B,H,DH,S]
    _Float16* O16  = (_Float16*)(ws + off); off += szX;        // [B*S, D]

    // fp32 -> fp16
    cvt_f32_f16<<<4096, 256, 0, stream>>>(hidden, Xh, Mc * Dc);
    cvt_f32_f16<<<2048, 256, 0, stream>>>(Wq, Wq16, Dc * Dc);
    cvt_f32_f16<<<2048, 256, 0, stream>>>(Wk, Wk16, Dc * Dc);
    cvt_f32_f16<<<2048, 256, 0, stream>>>(Wv, Wv16, Dc * Dc);
    cvt_f32_f16<<<2048, 256, 0, stream>>>(Wo, Wo16, Dc * Dc);

    // projections: (4096/32)*(2048/64) = 4096 waves -> 1024 blocks of 4 waves
    gemm_xwt<<<1024, 128, 0, stream>>>(Xh, Wq16, Qb, 0);
    gemm_xwt<<<1024, 128, 0, stream>>>(Xh, Wk16, Kb, 0);
    gemm_xwt<<<1024, 128, 0, stream>>>(Xh, Wv16, Vtb, 2);

    // flash attention: B*H*(S/16) = 4096 waves -> 1024 blocks
    flash_attn<<<1024, 128, 0, stream>>>(Qb, Kb, Vtb, amask, O16);

    // output projection -> f32 d_out
    gemm_xwt<<<1024, 128, 0, stream>>>(O16, Wo16, out, 3);
}
